// DocumentGraph_67954972557885
// MI455X (gfx1250) — compile-verified, matrix operand order checked by reading, not run on hardware
//
#include <hip/hip_runtime.h>

// HyperGAT (2 layers) for MI455X / gfx1250, wave32.
// - Heavy GEMMs (P@x, Q@edge, x@W) on v_wmma_f32_16x16x32_bf16, f32 accum.
// - LDS B panels stored pre-swizzled in WMMA fragment layout (b128 LDS reads).
// - Softmax kernels stage HT / score rows via global_load_async_to_lds_* (ASYNCcnt).
// - All score projections algebraically folded into per-feature vectors.

#define Bsz 32
#define Nn  1024
#define Ee  512
#define Dd  300
#define FP  320          // feature dim padded (multiple of 64 for K-loop)
#define NEGF (-9.0e15f)

typedef __attribute__((ext_vector_type(16))) __bf16 v16bf;
typedef __attribute__((ext_vector_type(8)))  float  v8f;
union FragB16 { v16bf v; unsigned short u[16]; uint4 q[2]; };

__device__ __forceinline__ unsigned short f2bf(float f) {
  unsigned int u = __float_as_uint(f);
  u += 0x7FFFu + ((u >> 16) & 1u);          // round-to-nearest-even
  return (unsigned short)(u >> 16);
}
__device__ __forceinline__ float wave_sum(float v) {
#pragma unroll
  for (int o = 16; o > 0; o >>= 1) v += __shfl_xor(v, o, 32);
  return v;
}
__device__ __forceinline__ float wave_max(float v) {
#pragma unroll
  for (int o = 16; o > 0; o >>= 1) v = fmaxf(v, __shfl_xor(v, o, 32));
  return v;
}

// ---------------------------------------------------------------------------
// Precompute fused attention vectors:
//   v1 = w2 @ a[Fo:], v2 = w2 @ a2[:Fo], v3 = w3 @ a2[Fo:], c = wc . a[:Fo]
// and pad w_2 -> bf16 [FP x FP].
// ---------------------------------------------------------------------------
__global__ void prep_kernel(const float* __restrict__ w2_1, const float* __restrict__ w3_1,
                            const float* __restrict__ w2_2, const float* __restrict__ w3_2,
                            const float* __restrict__ w_2,
                            const float* __restrict__ a_1,  const float* __restrict__ a2_1,
                            const float* __restrict__ a_2,  const float* __restrict__ a2_2,
                            const float* __restrict__ wc_1, const float* __restrict__ wc_2,
                            float* __restrict__ vbuf, float* __restrict__ cbuf,
                            unsigned short* __restrict__ wbf)
{
  const int bv = blockIdx.x;
  if (bv < 6) {
    const float* W; const float* av;
    switch (bv) {
      case 0: W = w2_1; av = a_1  + Dd; break;   // v1_1
      case 1: W = w2_1; av = a2_1;      break;   // v2_1
      case 2: W = w3_1; av = a2_1 + Dd; break;   // v3_1
      case 3: W = w2_2; av = a_2  + Dd; break;   // v1_2
      case 4: W = w2_2; av = a2_2;      break;   // v2_2
      default: W = w3_2; av = a2_2 + Dd; break;  // v3_2
    }
    const int i = threadIdx.x;
    if (i < FP) {
      float s = 0.f;
      if (i < Dd) for (int j = 0; j < Dd; ++j) s += W[(long long)i * Dd + j] * av[j];
      vbuf[(long long)bv * FP + i] = s;
    }
    if (bv == 0 && i == 0) { float c = 0.f; for (int j = 0; j < Dd; ++j) c += wc_1[j] * a_1[j]; cbuf[0] = c; }
    if (bv == 3 && i == 0) { float c = 0.f; for (int j = 0; j < Dd; ++j) c += wc_2[j] * a_2[j]; cbuf[1] = c; }
  } else {
    const int idx = (bv - 6) * blockDim.x + threadIdx.x;
    if (idx < FP * FP) {
      const int r = idx / FP, c = idx % FP;
      float v = (r < Dd && c < Dd) ? w_2[(long long)r * Dd + c] : 0.f;
      wbf[idx] = f2bf(v);
    }
  }
}

// Embedding gather: x[row] = emb[inputs[row]], zero-padded to FP, f32 + bf16.
__global__ void gather_kernel(const int* __restrict__ idx, const float* __restrict__ emb,
                              float* __restrict__ xf, unsigned short* __restrict__ xb)
{
  const int row = blockIdx.x;
  const float* er = emb + (long long)idx[row] * Dd;
  float* xr = xf + (long long)row * FP;
  unsigned short* br = xb + (long long)row * FP;
  for (int i = threadIdx.x; i < FP; i += blockDim.x) {
    const float v = (i < Dd) ? er[i] : 0.f;
    xr[i] = v; br[i] = f2bf(v);
  }
}

// Per-row fused scores: s1 = leaky(c + x.v1, alpha),  sn = x.v2  (one wave/row)
__global__ void scores_kernel(const float* __restrict__ x, const float* __restrict__ v1,
                              const float* __restrict__ v2, const float* __restrict__ cptr,
                              float alpha, float* __restrict__ s1, float* __restrict__ sn, int rows)
{
  const int w = (int)((blockIdx.x * (long long)blockDim.x + threadIdx.x) >> 5);
  const int lane = threadIdx.x & 31;
  if (w >= rows) return;
  const float* xr = x + (long long)w * FP;
  float d1 = 0.f, d2 = 0.f;
  for (int i = lane; i < FP; i += 32) { const float v = xr[i]; d1 += v * v1[i]; d2 += v * v2[i]; }
  d1 = wave_sum(d1); d2 = wave_sum(d2);
  if (lane == 0) {
    const float t = cptr[0] + d1;
    s1[w] = (t > 0.f) ? t : alpha * t;
    sn[w] = d2;
  }
}

// ---------------------------------------------------------------------------
// Edge attention: P[b,e,:] = softmax_n( mask ? s1[b,n] : NEG ) -> bf16.
// HT row (4KB) + s1 row (4KB) staged once via async DMA to LDS; masked scores
// cached in LDS so the 3 softmax passes touch global memory exactly once.
// ---------------------------------------------------------------------------
__global__ void edge_softmax_kernel(const int* __restrict__ HT, const float* __restrict__ s1,
                                    unsigned short* __restrict__ P)
{
  __shared__ int   htRow[Nn];      // reused as float masked-score row after pass 1
  __shared__ float s1Row[Nn];
  __shared__ float smax[4], ssum[4], sbc[2];
  const int be = blockIdx.x;                 // b*E + e
  const int b = be >> 9;
  const int* mp = HT + (long long)be * Nn;
  const float* sp = s1 + (long long)b * Nn;
  const int tid = threadIdx.x, lane = tid & 31, wid = tid >> 5;

  { // CDNA5 async global->LDS copy (ASYNCcnt)
    const unsigned htOff = (unsigned)(size_t)&htRow[0];
    const unsigned s1Off = (unsigned)(size_t)&s1Row[0];
    for (int i = tid; i < Nn / 4; i += 128) {
      const int*   g0 = mp + i * 4;
      const float* g1 = sp + i * 4;
      asm volatile("global_load_async_to_lds_b128 %0, %2, off\n\t"
                   "global_load_async_to_lds_b128 %1, %3, off"
                   :: "v"(htOff + i * 16u), "v"(s1Off + i * 16u), "v"(g0), "v"(g1)
                   : "memory");
    }
    asm volatile("s_wait_asynccnt 0x0" ::: "memory");
  }
  __syncthreads();

  float* valRow = (float*)htRow;
  float mx = NEGF;
  for (int n = tid; n < Nn; n += 128) {
    const float v = (htRow[n] != 0) ? s1Row[n] : NEGF;   // read-then-overwrite, same thread
    valRow[n] = v;
    mx = fmaxf(mx, v);
  }
  mx = wave_max(mx);
  if (lane == 0) smax[wid] = mx;
  __syncthreads();
  if (tid == 0) sbc[0] = fmaxf(fmaxf(smax[0], smax[1]), fmaxf(smax[2], smax[3]));
  __syncthreads();
  const float M = sbc[0];

  float z = 0.f;
  for (int n = tid; n < Nn; n += 128) z += __expf(valRow[n] - M);
  z = wave_sum(z);
  if (lane == 0) ssum[wid] = z;
  __syncthreads();
  if (tid == 0) sbc[1] = ssum[0] + ssum[1] + ssum[2] + ssum[3];
  __syncthreads();
  const float rz = 1.f / sbc[1];

  unsigned short* pr = P + (long long)be * Nn;
  for (int n = tid; n < Nn; n += 128) pr[n] = f2bf(__expf(valRow[n] - M) * rz);
}

// ---------------------------------------------------------------------------
// Node attention: Q[b,n,:] = softmax_e( mask ? leaky(sn[b,n]+se[b,e]) : NEG ).
// Strided HT column gathered once per block via per-lane async b32 DMA.
// ---------------------------------------------------------------------------
__global__ void node_softmax_kernel(const int* __restrict__ HT, const float* __restrict__ sn,
                                    const float* __restrict__ se, float alpha,
                                    unsigned short* __restrict__ Q)
{
  __shared__ int   htCol[Ee];      // reused as float masked-score row after pass 1
  __shared__ float seRow[Ee];
  __shared__ float smax[4], ssum[4], sbc[2];
  const int bn = blockIdx.x;                 // b*N + n
  const int b = bn >> 10;
  const int n = bn & (Nn - 1);
  const int* mp = HT + (long long)b * Ee * Nn + n;   // stride Nn over e
  const float* sep = se + (long long)b * Ee;
  const float snv = sn[bn];
  const int tid = threadIdx.x, lane = tid & 31, wid = tid >> 5;

  { // async gather of the strided HT column + contiguous se row
    for (int e = tid; e < Ee; e += 128) {
      const int* g = mp + (long long)e * Nn;
      asm volatile("global_load_async_to_lds_b32 %0, %1, off"
                   :: "v"((unsigned)(size_t)&htCol[e]), "v"(g) : "memory");
    }
    for (int i = tid; i < Ee / 4; i += 128) {
      const float* g = sep + i * 4;
      asm volatile("global_load_async_to_lds_b128 %0, %1, off"
                   :: "v"((unsigned)(size_t)&seRow[i * 4]), "v"(g) : "memory");
    }
    asm volatile("s_wait_asynccnt 0x0" ::: "memory");
  }
  __syncthreads();

  float* valRow = (float*)htCol;
  float mx = NEGF;
  for (int e = tid; e < Ee; e += 128) {
    float t = snv + seRow[e]; t = (t > 0.f) ? t : alpha * t;
    const float v = (htCol[e] != 0) ? t : NEGF;
    valRow[e] = v;
    mx = fmaxf(mx, v);
  }
  mx = wave_max(mx);
  if (lane == 0) smax[wid] = mx;
  __syncthreads();
  if (tid == 0) sbc[0] = fmaxf(fmaxf(smax[0], smax[1]), fmaxf(smax[2], smax[3]));
  __syncthreads();
  const float M = sbc[0];

  float z = 0.f;
  for (int e = tid; e < Ee; e += 128) z += __expf(valRow[e] - M);
  z = wave_sum(z);
  if (lane == 0) ssum[wid] = z;
  __syncthreads();
  if (tid == 0) sbc[1] = ssum[0] + ssum[1] + ssum[2] + ssum[3];
  __syncthreads();
  const float rz = 1.f / sbc[1];

  unsigned short* qr = Q + (long long)bn * Ee;
  for (int e = tid; e < Ee; e += 128) qr[e] = f2bf(__expf(valRow[e] - M) * rz);
}

// ---------------------------------------------------------------------------
// Batched bf16 GEMM: C[z] = A[z](MxK bf16,row-major) * B[z](K x >=colspan bf16).
// Block 256 thr / 8 waves: 128x32 tile, K-step 64 -> 4 wmma per barrier pair.
// LDS B panel stored in WMMA fragment layout: per-lane fragment is contiguous
// 32B -> 2x ds_load_b128. A fragments: 2x 16B global loads. B columns are
// zero-padded to FP so panel loads need no guards (no scalarization).
// Fragment layouts per CDNA5 ISA 7.12.2 (wave32).
// ---------------------------------------------------------------------------
__global__ void __launch_bounds__(256)
gemm_bf16_kernel(const unsigned short* __restrict__ A, long long lda, long long strideA,
                 const unsigned short* __restrict__ Bm, long long ldb, long long strideB,
                 float* __restrict__ C, long long ldc, long long strideC,
                 int M, int Kdim, int Ncols)
{
  // [p(2 k-subpanels)][c(2 col tiles)][lane(32)][slot(16)] u16 = 4KB
  __shared__ unsigned short Bs[2 * 2 * 32 * 16];
  const int tid  = threadIdx.x;
  const int wid  = tid >> 5;
  const int lane = tid & 31;
  const int half = lane >> 4;
  const int l16  = lane & 15;
  const int rowBase = blockIdx.y * 128 + wid * 16;
  const int colBase = blockIdx.x * 32;
  const unsigned short* Ab = A + (long long)blockIdx.z * strideA + (long long)(rowBase + l16) * lda;
  const unsigned short* Bb = Bm + (long long)blockIdx.z * strideB + colBase;
  float* Cb = C + (long long)blockIdx.z * strideC;

  // staging coords: thread covers row skk (0..63), cols scc0..scc0+7
  const int skk   = tid >> 2;
  const int scc0  = (tid & 3) * 8;
  const int sp    = skk >> 5;
  const int skk32 = skk & 31;
  const int shb   = (skk32 >> 3) & 1;
  const int sslot = (skk32 & 7) + 8 * (skk32 >> 4);
  const int sc    = scc0 >> 4;
  const int sl0   = scc0 & 15;
  const int sbase = ((sp * 2 + sc) * 32 + 16 * shb + sl0) * 16 + sslot;

  v8f acc0 = {0.f,0.f,0.f,0.f,0.f,0.f,0.f,0.f};
  v8f acc1 = {0.f,0.f,0.f,0.f,0.f,0.f,0.f,0.f};

  for (int k0 = 0; k0 < Kdim; k0 += 64) {
    // stage 64x32 panel: one 16B load per thread, scatter 8 u16 to frag layout
    const uint4 pkt = *(const uint4*)(Bb + (long long)(k0 + skk) * ldb + scc0);
    const unsigned short* ps = (const unsigned short*)&pkt;
#pragma unroll
    for (int j = 0; j < 8; ++j) Bs[sbase + j * 16] = ps[j];
    __syncthreads();

#pragma unroll
    for (int p = 0; p < 2; ++p) {
      FragB16 fa, fb0, fb1;
      const unsigned short* ap = Ab + k0 + p * 32 + half * 8;
      fa.q[0] = *(const uint4*)(ap);
      fa.q[1] = *(const uint4*)(ap + 16);
      const uint4* bp0 = (const uint4*)&Bs[((p * 2 + 0) * 32 + lane) * 16];
      const uint4* bp1 = (const uint4*)&Bs[((p * 2 + 1) * 32 + lane) * 16];
      fb0.q[0] = bp0[0]; fb0.q[1] = bp0[1];
      fb1.q[0] = bp1[0]; fb1.q[1] = bp1[1];
      acc0 = __builtin_amdgcn_wmma_f32_16x16x32_bf16(false, fa.v, false, fb0.v,
                                                     (short)0, acc0, false, false);
      acc1 = __builtin_amdgcn_wmma_f32_16x16x32_bf16(false, fa.v, false, fb1.v,
                                                     (short)0, acc1, false, false);
    }
    __syncthreads();
  }

  const int col0 = colBase + l16;
  const int col1 = colBase + 16 + l16;
#pragma unroll
  for (int r = 0; r < 8; ++r) {
    const int row = rowBase + half * 8 + r;       // C layout: VGPR r -> M = 8*half + r
    if (row < M) {
      if (col0 < Ncols) Cb[(long long)row * ldc + col0] = acc0[r];
      if (col1 < Ncols) Cb[(long long)row * ldc + col1] = acc1[r];
    }
  }
}

// Row convert f32->bf16 (zero-pad cols >= Dd), optional se[row] = row . v3
__global__ void rowconv_kernel(const float* __restrict__ src, unsigned short* __restrict__ dst,
                               const float* __restrict__ v3, float* __restrict__ se, int rows)
{
  const int w = (int)((blockIdx.x * (long long)blockDim.x + threadIdx.x) >> 5);
  const int lane = threadIdx.x & 31;
  if (w >= rows) return;
  const float* sr = src + (long long)w * FP;
  unsigned short* dr = dst + (long long)w * FP;
  float acc = 0.f;
  for (int i = lane; i < FP; i += 32) {
    const float v = (i < Dd) ? sr[i] : 0.f;
    dr[i] = f2bf(v);
    if (v3) acc += v * v3[i];
  }
  if (v3) { acc = wave_sum(acc); if (lane == 0) se[w] = acc; }
}

// In-place ELU on f32 buffer + bf16 mirror (zero pad cols)
__global__ void elu_kernel(float* __restrict__ xf, unsigned short* __restrict__ xb, long long total)
{
  const long long i = (long long)blockIdx.x * blockDim.x + threadIdx.x;
  if (i >= total) return;
  const int col = (int)(i % FP);
  float r = 0.f;
  if (col < Dd) { const float v = xf[i]; r = (v > 0.f) ? v : (__expf(v) - 1.f); }
  xf[i] = r;
  xb[i] = f2bf(r);
}

extern "C" void kernel_launch(void* const* d_in, const int* in_sizes, int n_in,
                              void* d_out, int out_size, void* d_ws, size_t ws_size,
                              hipStream_t stream) {
  (void)in_sizes; (void)n_in; (void)out_size; (void)ws_size;
  const int*   inputs = (const int*)d_in[0];
  const int*   HT     = (const int*)d_in[1];
  const float* emb    = (const float*)d_in[2];
  const float* w2_1   = (const float*)d_in[3];
  const float* w3_1   = (const float*)d_in[4];
  const float* a_1    = (const float*)d_in[5];
  const float* a2_1   = (const float*)d_in[6];
  const float* wc_1   = (const float*)d_in[7];
  const float* w_2    = (const float*)d_in[8];
  const float* w2_2   = (const float*)d_in[9];
  const float* w3_2   = (const float*)d_in[10];
  const float* a_2    = (const float*)d_in[11];
  const float* a2_2   = (const float*)d_in[12];
  const float* wc_2   = (const float*)d_in[13];
  float* out = (float*)d_out;
  char* ws = (char*)d_ws;

  const size_t oX0f = 0;
  const size_t oX0b = oX0f + (size_t)Bsz * Nn * FP * 4;
  const size_t oX1f = oX0b + (size_t)Bsz * Nn * FP * 2;
  const size_t oX1b = oX1f + (size_t)Bsz * Nn * FP * 4;
  const size_t oP   = oX1b + (size_t)Bsz * Nn * FP * 2;
  const size_t oQ   = oP   + (size_t)Bsz * Ee * Nn * 2;
  const size_t oEf  = oQ   + (size_t)Bsz * Nn * Ee * 2;
  const size_t oEb  = oEf  + (size_t)Bsz * Ee * FP * 4;
  const size_t oS1  = oEb  + (size_t)Bsz * Ee * FP * 2;
  const size_t oSn  = oS1  + (size_t)Bsz * Nn * 4;
  const size_t oSe  = oSn  + (size_t)Bsz * Nn * 4;
  const size_t oV   = oSe  + (size_t)Bsz * Ee * 4;
  const size_t oC   = oV   + (size_t)6 * FP * 4;
  const size_t oW   = oC   + 256;

  float* X0f = (float*)(ws + oX0f);
  unsigned short* X0b = (unsigned short*)(ws + oX0b);
  float* X1f = (float*)(ws + oX1f);
  unsigned short* X1b = (unsigned short*)(ws + oX1b);
  unsigned short* P = (unsigned short*)(ws + oP);
  unsigned short* Q = (unsigned short*)(ws + oQ);
  float* Ef = (float*)(ws + oEf);
  unsigned short* Eb = (unsigned short*)(ws + oEb);
  float* S1 = (float*)(ws + oS1);
  float* Sn = (float*)(ws + oSn);
  float* Se = (float*)(ws + oSe);
  float* Vb = (float*)(ws + oV);
  float* Cc = (float*)(ws + oC);
  unsigned short* Wb = (unsigned short*)(ws + oW);

  const int nColT = FP / 32;   // 10 column tiles of 32 (B padded; stores guarded)

  prep_kernel<<<6 + FP, FP, 0, stream>>>(w2_1, w3_1, w2_2, w3_2, w_2,
                                         a_1, a2_1, a_2, a2_2, wc_1, wc_2, Vb, Cc, Wb);
  gather_kernel<<<Bsz * Nn, 64, 0, stream>>>(inputs, emb, X0f, X0b);

  // ---------------- layer 1 (transfer=False, concat=True, alpha=0.1) --------
  scores_kernel<<<(Bsz * Nn) / 8, 256, 0, stream>>>(X0f, Vb + 0 * FP, Vb + 1 * FP, Cc + 0,
                                                    0.1f, S1, Sn, Bsz * Nn);
  edge_softmax_kernel<<<Bsz * Ee, 128, 0, stream>>>(HT, S1, P);
  gemm_bf16_kernel<<<dim3(nColT, Ee / 128, Bsz), 256, 0, stream>>>(
      P, Nn, (long long)Ee * Nn, X0b, FP, (long long)Nn * FP,
      Ef, FP, (long long)Ee * FP, Ee, Nn, Dd);
  rowconv_kernel<<<(Bsz * Ee) / 8, 256, 0, stream>>>(Ef, Eb, Vb + 2 * FP, Se, Bsz * Ee);
  node_softmax_kernel<<<Bsz * Nn, 128, 0, stream>>>(HT, Sn, Se, 0.1f, Q);
  gemm_bf16_kernel<<<dim3(nColT, Nn / 128, Bsz), 256, 0, stream>>>(
      Q, Ee, (long long)Nn * Ee, Eb, FP, (long long)Ee * FP,
      X1f, FP, (long long)Nn * FP, Nn, Ee, Dd);
  elu_kernel<<<(unsigned)(((long long)Bsz * Nn * FP + 255) / 256), 256, 0, stream>>>(
      X1f, X1b, (long long)Bsz * Nn * FP);

  // ---------------- layer 2 (transfer=True, concat=False, alpha=0.2) --------
  scores_kernel<<<(Bsz * Nn) / 8, 256, 0, stream>>>(X1f, Vb + 3 * FP, Vb + 4 * FP, Cc + 1,
                                                    0.2f, S1, Sn, Bsz * Nn);
  // xt = x1 @ w_2 (shared weights: strideB = 0); result reuses X0f/X0b
  gemm_bf16_kernel<<<dim3(nColT, Nn / 128, Bsz), 256, 0, stream>>>(
      X1b, FP, (long long)Nn * FP, Wb, FP, 0,
      X0f, FP, (long long)Nn * FP, Nn, FP, Dd);
  rowconv_kernel<<<(Bsz * Nn) / 8, 256, 0, stream>>>(X0f, X0b, nullptr, nullptr, Bsz * Nn);
  edge_softmax_kernel<<<Bsz * Ee, 128, 0, stream>>>(HT, S1, P);
  gemm_bf16_kernel<<<dim3(nColT, Ee / 128, Bsz), 256, 0, stream>>>(
      P, Nn, (long long)Ee * Nn, X0b, FP, (long long)Nn * FP,
      Ef, FP, (long long)Ee * FP, Ee, Nn, Dd);
  rowconv_kernel<<<(Bsz * Ee) / 8, 256, 0, stream>>>(Ef, Eb, Vb + 5 * FP, Se, Bsz * Ee);
  node_softmax_kernel<<<Bsz * Nn, 128, 0, stream>>>(HT, Sn, Se, 0.2f, Q);
  // final node GEMM writes straight into d_out (ldc = 300)
  gemm_bf16_kernel<<<dim3(nColT, Nn / 128, Bsz), 256, 0, stream>>>(
      Q, Ee, (long long)Nn * Ee, Eb, FP, (long long)Ee * FP,
      out, Dd, (long long)Nn * Dd, Nn, Ee, Dd);
}
